// CGCNN_13572096656012
// MI455X (gfx1250) — compile-verified, matrix operand order checked by reading, not run on hardware
//
#include <hip/hip_runtime.h>
#include <math.h>

#define NNODES  20000
#define NEDGES  200000
#define HDIM    256
#define IN_NODE 256
#define IN_EDGE 128
#define NGROUP  64
#define NLAYER  2
#define BN_EPSF 1e-5f
#define LPAD    8                      // bf16 elements of LDS row padding (16B)

typedef __attribute__((ext_vector_type(16))) __bf16 v16bf;
typedef __attribute__((ext_vector_type(8)))  __bf16 v8bf;
typedef __attribute__((ext_vector_type(8)))  float  v8f;

// ---------------------------------------------------------------------------
// bf16 fragment load (global or LDS): per-lane base p already includes
// row*ld + (lane>>4)*8.  16-bit A/B layout (ISA 7.12.2, 16x32):
//   elements 0..7  = K [half*8 .. half*8+8)
//   elements 8..15 = K [16+half*8 .. 16+half*8+8)
// Two b128 loads, zero VALU conversion work.
// ---------------------------------------------------------------------------
__device__ __forceinline__ v16bf load_frag(const __bf16* p) {
  const v8bf lo = *reinterpret_cast<const v8bf*>(p);
  const v8bf hi = *reinterpret_cast<const v8bf*>(p + 16);
  v16bf f;
#pragma unroll
  for (int i = 0; i < 8; ++i) { f[i] = lo[i]; f[i + 8] = hi[i]; }
  return f;
}

__device__ __forceinline__ v8f wmma_bf16(v16bf a, v16bf b, v8f c) {
  return __builtin_amdgcn_wmma_f32_16x16x32_bf16(false, a, false, b,
                                                 (short)0, c, false, false);
}

// Async global->LDS copy of one 16B chunk per lane (ASYNCcnt-tracked).
__device__ __forceinline__ void async_copy16(void* lds_dst, const void* gsrc) {
  unsigned int l = (unsigned int)(unsigned long long)lds_dst; // flat low bits = LDS addr
  asm volatile("global_load_async_to_lds_b128 %0, %1, off"
               :: "v"(l), "v"(gsrc) : "memory");
}
__device__ __forceinline__ void wait_async0() {
  asm volatile("s_wait_asynccnt 0x0" ::: "memory");
}

__device__ __forceinline__ float sigmoidf_(float x) {
  return 1.0f / (1.0f + __expf(-x));
}
__device__ __forceinline__ float softplusf_(float x) {
  return (x > 20.0f) ? x : log1pf(__expf(x));
}

// ---------------------------------------------------------------------------
// f32 -> bf16 conversion (one-shot producers for GEMM operands)
// ---------------------------------------------------------------------------
__global__ void cvt_bf16(const float* __restrict__ s, __bf16* __restrict__ d,
                         long n) {
  long i = ((long)blockIdx.x * 256 + threadIdx.x) * 4;
  if (i + 3 < n) {
    const float4 v = *reinterpret_cast<const float4*>(s + i);
    d[i]     = (__bf16)v.x;
    d[i + 1] = (__bf16)v.y;
    d[i + 2] = (__bf16)v.z;
    d[i + 3] = (__bf16)v.w;
  } else {
    for (; i < n; ++i) d[i] = (__bf16)s[i];
  }
}

// ---------------------------------------------------------------------------
// Y[M x 256] = X[M x K] @ W[256 x K]^T (+bias); bf16 in, f32 accum.
// 8 waves/block; all share the 64-col B tile -> staged once into LDS via
// async loads.  Each wave owns a 16x64 output strip (A frag reused x4).
// B fragments batched before the WMMA group so DS waits drain progressively.
// ---------------------------------------------------------------------------
__global__ void __launch_bounds__(256)
wmma_gemm(const __bf16* __restrict__ X, int ldx,
          const __bf16* __restrict__ Wt, int ldw,
          const float* __restrict__ bias,
          float* __restrict__ Yf, __bf16* __restrict__ Ybf,
          int M, int K) {
  __shared__ __bf16 shB[64 * (HDIM + LPAD)];      // 33.8 KB max (K=256)
  const int t     = threadIdx.x;
  const int lane  = t & 31;
  const int wid   = t >> 5;
  const int nbase = blockIdx.y * 64;
  const int pitch = K + LPAD;

  // --- stage B tile (64 rows x K) into LDS with async b128 copies ----------
  const int cshift = (K == 256) ? 5 : 4;          // chunks(16B) per row
  const int cmask  = (1 << cshift) - 1;
  const int total  = 64 << cshift;
  for (int idx = t; idx < total; idx += 256) {
    const int r = idx >> cshift;
    const int c = idx & cmask;
    async_copy16(&shB[r * pitch + c * 8],
                 Wt + (size_t)(nbase + r) * ldw + c * 8);
  }
  wait_async0();
  __syncthreads();

  const int mt = blockIdx.x * 8 + wid;            // 16-row tile index
  if (mt * 16 < M) {                              // whole-wave guard, EXEC all-1
    const int row  = lane & 15;
    const int half = lane >> 4;

    const __bf16* pA = X + (size_t)(mt * 16 + row) * ldx + half * 8;
    const __bf16* pB[4];
#pragma unroll
    for (int j = 0; j < 4; ++j)
      pB[j] = shB + (16 * j + row) * pitch + half * 8;

    v8f acc[4] = {};
    for (int k0 = 0; k0 < K; k0 += 32) {
      __builtin_prefetch(pA + k0 + 64, 0, 3);     // near-scope prefetch
      const v16bf a = load_frag(pA + k0);
      v16bf b[4];
#pragma unroll
      for (int j = 0; j < 4; ++j) b[j] = load_frag(pB[j] + k0);
#pragma unroll
      for (int j = 0; j < 4; ++j) acc[j] = wmma_bf16(a, b[j], acc[j]);
    }

#pragma unroll
    for (int j = 0; j < 4; ++j) {
      const int col = nbase + 16 * j + row;
      const float bv = bias ? bias[col] : 0.0f;
#pragma unroll
      for (int r = 0; r < 8; ++r) {               // C/D: VGPR r -> M = r / r+8
        const size_t o = (size_t)(mt * 16 + half * 8 + r) * HDIM + col;
        const float v = acc[j][r] + bv;
        if (Yf)  Yf[o]  = v;
        if (Ybf) Ybf[o] = (__bf16)v;
      }
    }
  }
}

// ---------------------------------------------------------------------------
// Fused edge kernel (16 edges x 32 channels per wave):
//   z_f = e@WfE^T + hiF[dst] + hjF[src]   (bias folded into hiF)
//   z_s = e@WsE^T + hiS[dst] + hjS[src]
//   atomicAdd(agg[dst], sigmoid(z_f)*softplus(z_s))
// All 8 waves share the 16-edge A tile -> async-staged into LDS once.
// z tiles live only in accumulators (no message tensor in memory).
// ---------------------------------------------------------------------------
__global__ void __launch_bounds__(256)
edge_msg_wmma(const __bf16* __restrict__ e,
              const __bf16* __restrict__ WfE, const __bf16* __restrict__ WsE,
              const float* __restrict__ hiF, const float* __restrict__ hjF,
              const float* __restrict__ hiS, const float* __restrict__ hjS,
              const int* __restrict__ srcI, const int* __restrict__ dstI,
              float* __restrict__ agg) {
  __shared__ __bf16 shA[16 * (HDIM + LPAD)];      // 8.4 KB
  const int t     = threadIdx.x;
  const int lane  = t & 31;
  const int wid   = t >> 5;
  const int ebase = blockIdx.x * 16;
  const int pitch = HDIM + LPAD;

  // --- stage A tile (16 edges x 256) into LDS: 512 x 16B chunks ------------
#pragma unroll
  for (int it = 0; it < 2; ++it) {
    const int idx = it * 256 + t;
    const int r = idx >> 5;                       // 32 chunks per row
    const int c = idx & 31;
    async_copy16(&shA[r * pitch + c * 8],
                 e + (size_t)(ebase + r) * HDIM + c * 8);
  }
  wait_async0();
  __syncthreads();

  const int nbase = wid * 32;                     // channel tile per wave
  const int row   = lane & 15;
  const int half  = lane >> 4;

  const __bf16* pA  = shA + row * pitch + half * 8;
  const __bf16* pF0 = WfE + (size_t)(nbase + row)      * (3 * HDIM) + half * 8;
  const __bf16* pF1 = WfE + (size_t)(nbase + 16 + row) * (3 * HDIM) + half * 8;
  const __bf16* pS0 = WsE + (size_t)(nbase + row)      * (3 * HDIM) + half * 8;
  const __bf16* pS1 = WsE + (size_t)(nbase + 16 + row) * (3 * HDIM) + half * 8;

  v8f aF0 = {}, aF1 = {}, aS0 = {}, aS1 = {};
  for (int k0 = 0; k0 < HDIM; k0 += 32) {
    const v16bf a  = load_frag(pA + k0);
    const v16bf b0 = load_frag(pF0 + k0);
    const v16bf b1 = load_frag(pF1 + k0);
    const v16bf b2 = load_frag(pS0 + k0);
    const v16bf b3 = load_frag(pS1 + k0);
    aF0 = wmma_bf16(a, b0, aF0);
    aF1 = wmma_bf16(a, b1, aF1);
    aS0 = wmma_bf16(a, b2, aS0);
    aS1 = wmma_bf16(a, b3, aS1);
  }

  const int c0 = nbase + row;
#pragma unroll
  for (int r = 0; r < 8; ++r) {
    const int m = ebase + half * 8 + r;           // edge id for VGPR r
    const int d = dstI[m];
    const int s = srcI[m];
    {
      const size_t id = (size_t)d * HDIM + c0;
      const size_t is = (size_t)s * HDIM + c0;
      const float zf = aF0[r] + hiF[id] + hjF[is];
      const float zs = aS0[r] + hiS[id] + hjS[is];
      atomicAdd(&agg[id], sigmoidf_(zf) * softplusf_(zs));
    }
    {
      const size_t id = (size_t)d * HDIM + c0 + 16;
      const size_t is = (size_t)s * HDIM + c0 + 16;
      const float zf = aF1[r] + hiF[id] + hjF[is];
      const float zs = aS1[r] + hiS[id] + hjS[is];
      atomicAdd(&agg[id], sigmoidf_(zf) * softplusf_(zs));
    }
  }
}

// ---------------------------------------------------------------------------
// Small helper kernels
// ---------------------------------------------------------------------------
__global__ void deg_kernel(const int* __restrict__ dstI, float* __restrict__ deg) {
  int i = blockIdx.x * 256 + threadIdx.x;
  if (i < NEDGES) atomicAdd(&deg[dstI[i]], 1.0f);
}

__global__ void invdeg_kernel(const float* __restrict__ deg, float* __restrict__ inv) {
  int i = blockIdx.x * 256 + threadIdx.x;
  if (i < NNODES) inv[i] = 1.0f / fmaxf(deg[i], 1.0f);
}

// h += agg*inv_deg; per-channel partial sum & sumsq (thread = channel).
__global__ void __launch_bounds__(256)
update_stats(float* __restrict__ h, const float* __restrict__ agg,
             const float* __restrict__ inv,
             float* __restrict__ ssum, float* __restrict__ ssq) {
  const int c  = threadIdx.x;
  const int r0 = blockIdx.x * 128;
  float s = 0.0f, q = 0.0f;
  for (int i = 0; i < 128; ++i) {
    const int r = r0 + i;
    if (r >= NNODES) break;
    const size_t idx = (size_t)r * HDIM + c;
    const float v = h[idx] + agg[idx] * inv[r];
    h[idx] = v;
    s += v;
    q += v * v;
  }
  atomicAdd(&ssum[c], s);
  atomicAdd(&ssq[c], q);
}

__global__ void __launch_bounds__(256)
bn_apply(float* __restrict__ h, __bf16* __restrict__ h_bf,
         const float* __restrict__ ssum, const float* __restrict__ ssq,
         const float* __restrict__ gam, const float* __restrict__ bet) {
  const int r = blockIdx.x;
  const int c = threadIdx.x;
  const float invN = 1.0f / (float)NNODES;
  const float mu  = ssum[c] * invN;
  const float var = ssq[c] * invN - mu * mu;
  const float rs  = rsqrtf(var + BN_EPSF);
  const size_t idx = (size_t)r * HDIM + c;
  const float v = (h[idx] - mu) * rs * gam[c] + bet[c];
  h[idx]    = v;
  h_bf[idx] = (__bf16)v;                          // feeds next layer's GEMMs
}

// batch is sorted: accumulate in registers, flush on group change.
__global__ void __launch_bounds__(256)
pool_kernel(const float* __restrict__ h, const int* __restrict__ batch,
            float* __restrict__ pool, float* __restrict__ gsz) {
  const int c  = threadIdx.x;
  const int r0 = blockIdx.x * 128;
  if (r0 >= NNODES) return;
  const int r1 = (r0 + 128 < NNODES) ? r0 + 128 : NNODES;
  int g = batch[r0];
  float acc = 0.0f;
  int cnt = 0;
  for (int r = r0; r < r1; ++r) {
    const int gr = batch[r];
    if (gr != g) {
      atomicAdd(&pool[(size_t)g * HDIM + c], acc);
      if (c == 0) atomicAdd(&gsz[g], (float)cnt);
      acc = 0.0f; cnt = 0; g = gr;
    }
    acc += h[(size_t)r * HDIM + c];
    ++cnt;
  }
  atomicAdd(&pool[(size_t)g * HDIM + c], acc);
  if (c == 0) atomicAdd(&gsz[g], (float)cnt);
}

__global__ void pool_norm(float* __restrict__ pool, const float* __restrict__ gsz) {
  const int g = blockIdx.x, c = threadIdx.x;
  const float d = fmaxf(gsz[g], 1.0f);
  const size_t i = (size_t)g * HDIM + c;
  pool[i] = softplusf_(pool[i] / d);
}

__global__ void final_kernel(const float* __restrict__ pool,
                             const float* __restrict__ W_fc,
                             const float* __restrict__ b_fc,
                             float* __restrict__ out) {
  const int g = blockIdx.x, o = threadIdx.x;
  const float* pr = pool + (size_t)g * HDIM;
  const float* wr = W_fc + (size_t)o * HDIM;
  float acc = b_fc[o];
  for (int c = 0; c < HDIM; ++c) acc += pr[c] * wr[c];
  out[(size_t)g * HDIM + o] = softplusf_(acc);
}

// ---------------------------------------------------------------------------
extern "C" void kernel_launch(void* const* d_in, const int* in_sizes, int n_in,
                              void* d_out, int out_size, void* d_ws, size_t ws_size,
                              hipStream_t stream) {
  (void)in_sizes; (void)n_in; (void)out_size; (void)ws_size;
  const float* x      = (const float*)d_in[0];
  const int*   ei     = (const int*)  d_in[1];
  const float* eattr  = (const float*)d_in[2];
  const int*   batch  = (const int*)  d_in[3];
  const float* W_emb  = (const float*)d_in[4];
  const float* b_emb  = (const float*)d_in[5];
  const float* W_edge = (const float*)d_in[6];
  const float* b_edge = (const float*)d_in[7];
  const float* Wf     = (const float*)d_in[8];
  const float* bf     = (const float*)d_in[9];
  const float* Ws     = (const float*)d_in[10];
  const float* bs     = (const float*)d_in[11];
  const float* gamma  = (const float*)d_in[12];
  const float* beta   = (const float*)d_in[13];
  const float* W_fc   = (const float*)d_in[14];
  const float* b_fc   = (const float*)d_in[15];
  float* out = (float*)d_out;

  const size_t NH = (size_t)NNODES * HDIM;
  const size_t EH = (size_t)NEDGES * HDIM;
  const size_t EA = (size_t)NEDGES * IN_EDGE;
  const size_t WF = (size_t)NLAYER * HDIM * 3 * HDIM;

  char* p = (char*)d_ws;
  float* h    = (float*)p; p += NH * 4;
  float* hiF  = (float*)p; p += NH * 4;
  float* hjF  = (float*)p; p += NH * 4;
  float* hiS  = (float*)p; p += NH * 4;
  float* hjS  = (float*)p; p += NH * 4;
  float* agg  = (float*)p; p += NH * 4;
  float* deg  = (float*)p; p += NNODES * 4;
  float* invd = (float*)p; p += NNODES * 4;
  float* ssum = (float*)p; p += HDIM * 4;
  float* ssq  = (float*)p; p += HDIM * 4;
  float* pool = (float*)p; p += (size_t)NGROUP * HDIM * 4;
  float* gsz  = (float*)p; p += NGROUP * 4;
  __bf16* x_bf     = (__bf16*)p; p += NH * 2;          // N*IN_NODE == NH
  __bf16* ea_bf    = (__bf16*)p; p += EA * 2;
  __bf16* ef_bf    = (__bf16*)p; p += EH * 2;
  __bf16* h_bf     = (__bf16*)p; p += NH * 2;
  __bf16* Wemb_bf  = (__bf16*)p; p += (size_t)HDIM * IN_NODE * 2;
  __bf16* Wedge_bf = (__bf16*)p; p += (size_t)HDIM * IN_EDGE * 2;
  __bf16* Wf_bf    = (__bf16*)p; p += WF * 2;
  __bf16* Ws_bf    = (__bf16*)p; p += WF * 2;

  const int* srcI = ei;
  const int* dstI = ei + NEDGES;

  const dim3 blk(256);
  const int  mtN = (NNODES / 16 + 7) / 8;   // 157 blocks for node GEMMs
  const int  mtE = (NEDGES / 16 + 7) / 8;   // 1563 blocks for edge embedding
  auto cvtg = [](size_t n) { return (unsigned)((n / 4 + 255) / 256); };

  // one-shot bf16 conversions of all GEMM operands
  cvt_bf16<<<cvtg(NH), blk, 0, stream>>>(x, x_bf, (long)NH);
  cvt_bf16<<<cvtg(EA), blk, 0, stream>>>(eattr, ea_bf, (long)EA);
  cvt_bf16<<<cvtg((size_t)HDIM * IN_NODE), blk, 0, stream>>>(W_emb, Wemb_bf,
                                                             HDIM * IN_NODE);
  cvt_bf16<<<cvtg((size_t)HDIM * IN_EDGE), blk, 0, stream>>>(W_edge, Wedge_bf,
                                                             HDIM * IN_EDGE);
  cvt_bf16<<<cvtg(WF), blk, 0, stream>>>(Wf, Wf_bf, (long)WF);
  cvt_bf16<<<cvtg(WF), blk, 0, stream>>>(Ws, Ws_bf, (long)WF);

  // degrees
  hipMemsetAsync(deg, 0, NNODES * sizeof(float), stream);
  deg_kernel<<<(NEDGES + 255) / 256, blk, 0, stream>>>(dstI, deg);
  invdeg_kernel<<<(NNODES + 255) / 256, blk, 0, stream>>>(deg, invd);

  // embeddings: h (f32+bf16), ef (bf16 only — sole consumer is edge WMMA)
  wmma_gemm<<<dim3(mtN, 4), blk, 0, stream>>>(x_bf, IN_NODE, Wemb_bf, IN_NODE,
                                              b_emb, h, h_bf, NNODES, IN_NODE);
  wmma_gemm<<<dim3(mtE, 4), blk, 0, stream>>>(ea_bf, IN_EDGE, Wedge_bf, IN_EDGE,
                                              b_edge, nullptr, ef_bf, NEDGES,
                                              IN_EDGE);

  for (int l = 0; l < NLAYER; ++l) {
    const __bf16* Wfl = Wf_bf + (size_t)l * HDIM * 3 * HDIM;
    const __bf16* Wsl = Ws_bf + (size_t)l * HDIM * 3 * HDIM;
    // node-level gate projections (bias folded into dst term)
    wmma_gemm<<<dim3(mtN, 4), blk, 0, stream>>>(h_bf, HDIM, Wfl,        3 * HDIM,
                                                bf + l * HDIM, hiF, nullptr,
                                                NNODES, HDIM);
    wmma_gemm<<<dim3(mtN, 4), blk, 0, stream>>>(h_bf, HDIM, Wfl + HDIM, 3 * HDIM,
                                                nullptr,       hjF, nullptr,
                                                NNODES, HDIM);
    wmma_gemm<<<dim3(mtN, 4), blk, 0, stream>>>(h_bf, HDIM, Wsl,        3 * HDIM,
                                                bs + l * HDIM, hiS, nullptr,
                                                NNODES, HDIM);
    wmma_gemm<<<dim3(mtN, 4), blk, 0, stream>>>(h_bf, HDIM, Wsl + HDIM, 3 * HDIM,
                                                nullptr,       hjS, nullptr,
                                                NNODES, HDIM);
    // fused edge GEMM + gating + scatter
    hipMemsetAsync(agg, 0, NH * sizeof(float), stream);
    edge_msg_wmma<<<NEDGES / 16, blk, 0, stream>>>(
        ef_bf, Wfl + 2 * HDIM, Wsl + 2 * HDIM, hiF, hjF, hiS, hjS,
        srcI, dstI, agg);
    // residual update + batchnorm
    hipMemsetAsync(ssum, 0, 2 * HDIM * sizeof(float), stream);
    update_stats<<<mtN, blk, 0, stream>>>(h, agg, invd, ssum, ssq);
    bn_apply<<<NNODES, blk, 0, stream>>>(h, h_bf, ssum, ssq, gamma + l * HDIM,
                                         beta + l * HDIM);
  }

  // pooling + FC head
  hipMemsetAsync(pool, 0, ((size_t)NGROUP * HDIM + NGROUP) * sizeof(float),
                 stream);
  pool_kernel<<<mtN, blk, 0, stream>>>(h, batch, pool, gsz);
  pool_norm<<<NGROUP, blk, 0, stream>>>(pool, gsz);
  final_kernel<<<NGROUP, blk, 0, stream>>>(pool, W_fc, b_fc, out);
}
